// SelfAttention_16063177687574
// MI455X (gfx1250) — matrix operand order repaired and checked by reading, not compiled, not run on hardware
//
#include <hip/hip_runtime.h>
#include <hip/hip_bf16.h>
#include <math.h>

// ---------------------------------------------------------------------------
// CDNA5 (gfx1250, wave32) WMMA helpers: v_wmma_f32_16x16x32_bf16
// ---------------------------------------------------------------------------
typedef __attribute__((ext_vector_type(16))) __bf16 bf16x16;
typedef __attribute__((ext_vector_type(8)))  __bf16 bf16x8;
typedef __attribute__((ext_vector_type(8)))  float  f32x8;
typedef __attribute__((ext_vector_type(4)))  float  f32x4;
typedef __attribute__((ext_vector_type(4)))  unsigned u32x4;
typedef __attribute__((ext_vector_type(8)))  int   i32x8;
typedef __attribute__((ext_vector_type(4)))  int   i32x4;

__device__ __forceinline__ bf16x16 combine8(bf16x8 lo, bf16x8 hi) {
    bf16x16 r;
#pragma unroll
    for (int i = 0; i < 8; i++) { r[i] = lo[i]; r[i + 8] = hi[i]; }
    return r;
}

__device__ __forceinline__ f32x8 wmma_bf16(bf16x16 a, bf16x16 b, f32x8 c) {
    return __builtin_amdgcn_wmma_f32_16x16x32_bf16(
        false, a, false, b, (short)0, c, false, false);
}

// ---------------------------------------------------------------------------
// Async DMA global->LDS (16B per lane), ASYNCcnt-tracked (CDNA5 ISA)
// ---------------------------------------------------------------------------
__device__ __forceinline__ void async_copy16(void* lds, const void* gptr) {
    unsigned l = (unsigned)(size_t)lds;        // flat addr low 32 = LDS offset
    asm volatile("global_load_async_to_lds_b128 %0, %1, off"
                 :: "v"(l), "v"(gptr) : "memory");
}
__device__ __forceinline__ void wait_async_le3() { asm volatile("s_wait_asynccnt 0x3" ::: "memory"); }
__device__ __forceinline__ void wait_async_le8() { asm volatile("s_wait_asynccnt 0x8" ::: "memory"); }
__device__ __forceinline__ void wait_async_0()   { asm volatile("s_wait_asynccnt 0x0" ::: "memory"); }

// ---------------------------------------------------------------------------
// Tensor Data Mover: 2D tile DMA global->LDS, TENSORcnt-tracked
// D# layout per CDNA5 ISA (group0: count/lds/global/type,
// group1: data_size=1 (2B), tensor_dim0/1, tile_dim0/1, dim0 stride).
// Toolchain here is the 6-arg builtin (clang-23 / therock-10.0 headers).
// ---------------------------------------------------------------------------
#if defined(__has_builtin)
#if __has_builtin(__builtin_amdgcn_tensor_load_to_lds)
#define HAVE_TDM 1
#endif
#endif
#ifndef HAVE_TDM
#define HAVE_TDM 0
#endif

#if HAVE_TDM
__device__ __forceinline__ void tdm_load_2d(void* lds, const void* gptr,
                                            unsigned tileX, unsigned tileY,
                                            unsigned strideElems)
{
    unsigned long long ga = (unsigned long long)(size_t)gptr;
    u32x4 g0;
    g0[0] = 1u;                                   // count=1 (valid), user mode
    g0[1] = (unsigned)(size_t)lds;                // lds_addr  (bits 63:32)
    g0[2] = (unsigned)(ga & 0xffffffffu);         // global_addr[31:0]  (95:64)
    g0[3] = (unsigned)((ga >> 32) & 0x1ffffffu)   // global_addr[56:32] (120:96)
          | (2u << 30);                           // type=2 "image"     (127:126)

    unsigned td0 = tileX;                         // tensor_dim0 = line length
    unsigned td1 = 0x100000u;                     // tensor_dim1: large (no OOB)
    i32x8 g1;
    g1[0] = (int)(1u << 16);                                  // data_size=1 (2B)
    g1[1] = (int)((td0 & 0xffffu) << 16);                     // tensor_dim0[15:0] @48
    g1[2] = (int)(((td0 >> 16) & 0xffffu) | ((td1 & 0xffffu) << 16));
    g1[3] = (int)(((td1 >> 16) & 0xffffu) | ((tileX & 0xffffu) << 16)); // tile_dim0 @112
    g1[4] = (int)(tileY & 0xffffu);                           // tile_dim1 @128, tile_dim2=0
    g1[5] = (int)strideElems;                                 // tensor_dim0_stride[31:0]
    g1[6] = 0;
    g1[7] = 0;
    i32x4 z4 = {0, 0, 0, 0};
    i32x8 z8 = {0, 0, 0, 0, 0, 0, 0, 0};
    __builtin_amdgcn_tensor_load_to_lds(g0, g1, z4, z4, z8, 0);
}
#endif

// ---------------------------------------------------------------------------
// f32 -> bf16 conversion (X) and transposed weight conversion W[K][C]->Wt[C][K]
// ---------------------------------------------------------------------------
__global__ void cvt_bf16(const float* __restrict__ in, __bf16* __restrict__ out, int n4)
{
    int i = blockIdx.x * blockDim.x + threadIdx.x;
    if (i >= n4) return;
    f32x4 v = *(const f32x4*)(in + i * 4);
#pragma unroll
    for (int j = 0; j < 4; j++) out[i * 4 + j] = (__bf16)v[j];
}

__global__ __launch_bounds__(256)
void transpose_bf16(const float* __restrict__ W, __bf16* __restrict__ Wt, int K, int C)
{
    __shared__ float t[32][33];
    const int c0 = blockIdx.x * 32, k0 = blockIdx.y * 32;
    const int tx = threadIdx.x & 31, ty = threadIdx.x >> 5;
#pragma unroll
    for (int i = ty; i < 32; i += 8) t[i][tx] = W[(size_t)(k0 + i) * C + c0 + tx];
    __syncthreads();
#pragma unroll
    for (int i = ty; i < 32; i += 8) Wt[(size_t)(c0 + i) * K + k0 + tx] = (__bf16)t[tx][i];
}

// ---------------------------------------------------------------------------
// GEMM: C[M,Ncols] = A[M,Kd] * Bt[Ncols,Kd]^T  (bf16 operands, f32 accum)
// Double-buffered LDS, async-DMA staged, DMA(i+1) overlaps WMMA(i).
// OUTMODE: 0 = f32 [M][Ncols], 1 = bf16 [M][Ncols], 2 = bf16 transposed [Ncols][M]
// ---------------------------------------------------------------------------
#define TBM 64
#define TBN 128
#define TBK 32

template<int OUTMODE>
__global__ __launch_bounds__(256)
void gemm_wmma(const __bf16* __restrict__ A, const __bf16* __restrict__ Bt,
               void* __restrict__ Cout, int M, int Kd, int Ncols)
{
    __shared__ __bf16 sA[2][TBM][TBK];     // 2 x 4 KB
    __shared__ __bf16 sB[2][TBN][TBK];     // 2 x 8 KB

    const int tid  = threadIdx.x;
    const int lane = tid & 31, wave = tid >> 5;
    const int l16  = lane & 15, half = lane >> 4;
    const int rowBase = blockIdx.y * TBM;
    const int colBase = blockIdx.x * TBN;
    const int wm = wave >> 2, wn = wave & 3;

    const int ar = tid >> 2, ac = (tid & 3) * 8;
    const int br = tid >> 1, bc = (tid & 1) * 16;
    const __bf16* Ag = A  + (size_t)(rowBase + ar) * Kd + ac;
    const __bf16* Bg = Bt + (size_t)(colBase + br) * Kd + bc;

    f32x8 acc[2][2] = {};
    const int nK = Kd / TBK;

    async_copy16(&sA[0][ar][ac],     Ag);
    async_copy16(&sB[0][br][bc],     Bg);
    async_copy16(&sB[0][br][bc + 8], Bg + 8);

    for (int i = 0; i < nK; i++) {
        const int cur  = i & 1;
        const bool more = (i + 1) < nK;
        if (more) {
            const int kk = (i + 1) * TBK;
            async_copy16(&sA[cur ^ 1][ar][ac],     Ag + kk);
            async_copy16(&sB[cur ^ 1][br][bc],     Bg + kk);
            async_copy16(&sB[cur ^ 1][br][bc + 8], Bg + kk + 8);
            wait_async_le3();
        } else {
            wait_async_0();
        }
        if (i + 2 < nK) {
            __builtin_prefetch(Ag + (i + 2) * TBK, 0, 0);
            __builtin_prefetch(Bg + (i + 2) * TBK, 0, 0);
        }
        __syncthreads();

        bf16x16 af[2], bfr[2];
#pragma unroll
        for (int mi = 0; mi < 2; mi++) {
            int r  = wm * 32 + mi * 16 + l16;
            int k0 = half * 8;
            af[mi] = combine8(*(const bf16x8*)&sA[cur][r][k0],
                              *(const bf16x8*)&sA[cur][r][k0 + 16]);
        }
#pragma unroll
        for (int ni = 0; ni < 2; ni++) {
            int c  = wn * 32 + ni * 16 + l16;
            int k0 = half * 8;
            bfr[ni] = combine8(*(const bf16x8*)&sB[cur][c][k0],
                               *(const bf16x8*)&sB[cur][c][k0 + 16]);
        }
#pragma unroll
        for (int mi = 0; mi < 2; mi++)
#pragma unroll
            for (int ni = 0; ni < 2; ni++)
                acc[mi][ni] = wmma_bf16(af[mi], bfr[ni], acc[mi][ni]);

        __syncthreads();
    }

    if constexpr (OUTMODE == 0) {
        float* C = (float*)Cout;
#pragma unroll
        for (int mi = 0; mi < 2; mi++)
#pragma unroll
            for (int ni = 0; ni < 2; ni++)
#pragma unroll
                for (int j = 0; j < 8; j++) {
                    int r  = rowBase + wm * 32 + mi * 16 + half * 8 + j;
                    int cc = colBase + wn * 32 + ni * 16 + l16;
                    C[(size_t)r * Ncols + cc] = acc[mi][ni][j];
                }
    } else if constexpr (OUTMODE == 1) {
        __bf16* C = (__bf16*)Cout;
#pragma unroll
        for (int mi = 0; mi < 2; mi++)
#pragma unroll
            for (int ni = 0; ni < 2; ni++)
#pragma unroll
                for (int j = 0; j < 8; j++) {
                    int r  = rowBase + wm * 32 + mi * 16 + half * 8 + j;
                    int cc = colBase + wn * 32 + ni * 16 + l16;
                    C[(size_t)r * Ncols + cc] = (__bf16)acc[mi][ni][j];
                }
    } else {
        __bf16* C = (__bf16*)Cout;          // [Ncols][M]; rows contiguous -> 16B packs
#pragma unroll
        for (int mi = 0; mi < 2; mi++)
#pragma unroll
            for (int ni = 0; ni < 2; ni++) {
                bf16x8 p;
#pragma unroll
                for (int j = 0; j < 8; j++) p[j] = (__bf16)acc[mi][ni][j];
                int cc = colBase + wn * 32 + ni * 16 + l16;
                int r0 = rowBase + wm * 32 + mi * 16 + half * 8;
                *(bf16x8*)&C[(size_t)cc * M + r0] = p;
            }
    }
}

// ---------------------------------------------------------------------------
// RoPE in-place on bf16 [B*T, NH, 128]; pos [B*T] int32
// ---------------------------------------------------------------------------
__global__ void rope_kernel(__bf16* __restrict__ buf, const int* __restrict__ pos,
                            int BT, int NH)
{
    int idx = blockIdx.x * blockDim.x + threadIdx.x;
    int total = BT * NH * 64;
    if (idx >= total) return;
    int h    = idx & 63;
    int t2   = idx >> 6;
    int head = t2 % NH;
    int bt   = t2 / NH;
    float p    = (float)pos[bt];
    float finv = __expf(-(float)h * 0.14391156831f);   // ln(10000)/64
    float s, c;
    __sincosf(p * finv, &s, &c);
    __bf16* base = buf + ((size_t)bt * NH + head) * 128;
    float x1 = (float)base[h], x2 = (float)base[h + 64];
    base[h]      = (__bf16)(x1 * c - x2 * s);
    base[h + 64] = (__bf16)(x2 * c + x1 * s);
}

// ---------------------------------------------------------------------------
// Causal flash attention (scale 1.0), GQA kv-head = n / (N/KH).
// 128 threads = 4 waves, 64 q-rows/block.  K/V tiles staged by the Tensor
// Data Mover (wave 0 issues 2 descriptors/tile, TENSORcnt pipeline), with
// per-lane async-DMA fallback.  Double-buffered: DMA(kt+1) overlaps kt.
// ---------------------------------------------------------------------------
#define QT 64
#define KT 32
#define HD 128

__global__ __launch_bounds__(128)
void flash_attn(const __bf16* __restrict__ Q,    // [BT, N, H]
                const __bf16* __restrict__ K,    // [BT, KH, H]
                const __bf16* __restrict__ Vt,   // [KH*H, BT]
                __bf16* __restrict__ O,          // [BT, N, H]
                int Bn, int T, int N, int KH)
{
    __shared__ __bf16 sK[2][KT][HD];      // 2 x 8 KB  [s][h]
    __shared__ __bf16 sV[2][HD][KT];      // 2 x 8 KB  [h][s]
    __shared__ __bf16 sP[4][16][KT];      // 4 KB      per-wave P staging

    const int tid  = threadIdx.x;
    const int lane = tid & 31, wave = tid >> 5;
    const int l16  = lane & 15, half = lane >> 4;
    const int qBase = blockIdx.x * QT;
    const int n = blockIdx.y, b = blockIdx.z;
    const int kh = n / (N / KH);
    const int qRow0 = qBase + wave * 16;
    const int BT = Bn * T;

#if HAVE_TDM
    auto stage = [&](int buf, int kt) {            // wave 0 only
        const int sb = kt * KT;
        tdm_load_2d(&sK[buf][0][0],
                    K + ((size_t)(b * T + sb) * KH + kh) * HD,
                    /*tileX=*/HD, /*tileY=*/KT, /*stride=*/KH * HD);
        tdm_load_2d(&sV[buf][0][0],
                    Vt + (size_t)kh * HD * BT + b * T + sb,
                    /*tileX=*/KT, /*tileY=*/HD, /*stride=*/BT);
    };
#else
    const int ks = tid >> 2, kc = (tid & 3) * 32;
    const __bf16* Kg = K + ((size_t)(b * T + ks) * KH + kh) * HD + kc;
    const __bf16* Vg = Vt + (size_t)(kh * HD + tid) * BT + b * T;
    auto stage = [&](int buf, int kt) {
        const int sb = kt * KT;
#pragma unroll
        for (int i = 0; i < 4; i++)
            async_copy16(&sK[buf][ks][kc + i * 8], Kg + (size_t)sb * KH * HD + i * 8);
#pragma unroll
        for (int i = 0; i < 4; i++)
            async_copy16(&sV[buf][tid][i * 8], Vg + sb + i * 8);
    };
#endif

    // Q fragments: 16 rows x 128, 4 K-steps (A-matrix layout)
    bf16x16 qfrag[4];
    {
        const __bf16* qptr = Q + ((size_t)(b * T + qRow0 + l16) * N + n) * HD;
#pragma unroll
        for (int c = 0; c < 4; c++) {
            int k0 = c * 32 + half * 8;
            qfrag[c] = combine8(*(const bf16x8*)(qptr + k0),
                                *(const bf16x8*)(qptr + k0 + 16));
        }
    }

    f32x8 o[8] = {};
    float m[8], l[8];
#pragma unroll
    for (int j = 0; j < 8; j++) { m[j] = -3.0e38f; l[j] = 0.0f; }

    const int waveRowMax = qRow0 + 15;
    const int nTiles = (qBase + QT + KT - 1) / KT;

#if HAVE_TDM
    if (wave == 0) stage(0, 0);
#else
    stage(0, 0);
#endif

    for (int kt = 0; kt < nTiles; kt++) {
        const int sBase = kt * KT;
        const int cur = kt & 1;
        const bool more = (kt + 1) < nTiles;
#if HAVE_TDM
        if (wave == 0) {
            if (more) { stage(cur ^ 1, kt + 1); __builtin_amdgcn_s_wait_tensorcnt(2); }
            else      { __builtin_amdgcn_s_wait_tensorcnt(0); }
        }
#else
        if (more) { stage(cur ^ 1, kt + 1); wait_async_le8(); }
        else      { wait_async_0(); }
#endif
        __syncthreads();

        if (sBase <= waveRowMax) {
            // --- S = Q K^T (batch 4 fragment loads, then 4 WMMAs) ---
            f32x8 sc[2] = {};
#pragma unroll
            for (int ni = 0; ni < 2; ni++) {
                int key = ni * 16 + l16;
                bf16x16 kf[4];
#pragma unroll
                for (int c = 0; c < 4; c++) {
                    int k0 = c * 32 + half * 8;
                    kf[c] = combine8(*(const bf16x8*)&sK[cur][key][k0],
                                     *(const bf16x8*)&sK[cur][key][k0 + 16]);
                }
#pragma unroll
                for (int c = 0; c < 4; c++)
                    sc[ni] = wmma_bf16(qfrag[c], kf[c], sc[ni]);
            }
            // --- causal mask ---
#pragma unroll
            for (int ni = 0; ni < 2; ni++) {
                int sGlob = sBase + ni * 16 + l16;
#pragma unroll
                for (int j = 0; j < 8; j++)
                    if (sGlob > qRow0 + half * 8 + j) sc[ni][j] = -3.0e38f;
            }
            // --- online softmax (reductions across 16-lane halves) ---
            float alpha[8];
#pragma unroll
            for (int j = 0; j < 8; j++) {
                float t = fmaxf(sc[0][j], sc[1][j]);
#pragma unroll
                for (int d = 1; d < 16; d <<= 1)
                    t = fmaxf(t, __shfl_xor(t, d, 32));
                float mnew = fmaxf(m[j], t);
                alpha[j] = __expf(m[j] - mnew);
                m[j] = mnew;
            }
#pragma unroll
            for (int j = 0; j < 8; j++) {
                float p0 = __expf(sc[0][j] - m[j]);
                float p1 = __expf(sc[1][j] - m[j]);
                sc[0][j] = p0; sc[1][j] = p1;
                float t = p0 + p1;
#pragma unroll
                for (int d = 1; d < 16; d <<= 1)
                    t += __shfl_xor(t, d, 32);
                l[j] = l[j] * alpha[j] + t;
            }
#pragma unroll
            for (int c = 0; c < 8; c++)
#pragma unroll
                for (int j = 0; j < 8; j++) o[c][j] *= alpha[j];

            // --- relayout P (C/D -> A layout) via per-wave LDS ---
#pragma unroll
            for (int ni = 0; ni < 2; ni++)
#pragma unroll
                for (int j = 0; j < 8; j++)
                    sP[wave][half * 8 + j][ni * 16 + l16] = (__bf16)sc[ni][j];
            asm volatile("s_wait_dscnt 0x0" ::: "memory");

            bf16x16 pfrag = combine8(*(const bf16x8*)&sP[wave][l16][half * 8],
                                     *(const bf16x8*)&sP[wave][l16][half * 8 + 16]);

            // --- O += P V (batch 4 fragment loads, then 4 WMMAs) ---
#pragma unroll
            for (int g = 0; g < 2; g++) {
                bf16x16 vf[4];
#pragma unroll
                for (int c = 0; c < 4; c++) {
                    int hcol = (g * 4 + c) * 16 + l16;
                    int k0   = half * 8;
                    vf[c] = combine8(*(const bf16x8*)&sV[cur][hcol][k0],
                                     *(const bf16x8*)&sV[cur][hcol][k0 + 16]);
                }
#pragma unroll
                for (int c = 0; c < 4; c++)
                    o[g * 4 + c] = wmma_bf16(pfrag, vf[c], o[g * 4 + c]);
            }
        }
        __syncthreads();
    }

    // --- normalize, store bf16 [BT, N, H] ---
    float inv[8];
#pragma unroll
    for (int j = 0; j < 8; j++) inv[j] = 1.0f / l[j];
#pragma unroll
    for (int c = 0; c < 8; c++)
#pragma unroll
        for (int j = 0; j < 8; j++) {
            int qr = qRow0 + half * 8 + j;
            int h  = c * 16 + l16;
            O[((size_t)(b * T + qr) * N + n) * HD + h] = (__bf16)(o[c][j] * inv[j]);
        }
}

// ---------------------------------------------------------------------------
// Launch: cvt/transpose -> QKV GEMMs -> RoPE -> flash attn -> out GEMM
// ---------------------------------------------------------------------------
extern "C" void kernel_launch(void* const* d_in, const int* in_sizes, int n_in,
                              void* d_out, int out_size, void* d_ws, size_t ws_size,
                              hipStream_t stream)
{
    const float* X    = (const float*)d_in[0];
    const int*   qpos = (const int*)  d_in[1];
    const float* Wq   = (const float*)d_in[2];
    const float* Wk   = (const float*)d_in[3];
    const float* Wv   = (const float*)d_in[4];
    const float* Wo   = (const float*)d_in[5];
    float* out = (float*)d_out;

    const int Bn = 2, T = 2048, D = 2048, N = 16, KH = 4, H = HD;
    const int M = Bn * T;                         // 4096

    __bf16* Xb  = (__bf16*)d_ws;                  // [M][D]
    __bf16* Qb  = Xb  + (size_t)M * D;            // [M][N*H]
    __bf16* Kb  = Qb  + (size_t)M * N * H;        // [M][KH*H]
    __bf16* Vt  = Kb  + (size_t)M * KH * H;       // [KH*H][M]  (transposed)
    __bf16* Ab  = Vt  + (size_t)M * KH * H;       // [M][N*H]
    __bf16* Wqt = Ab  + (size_t)M * N * H;        // [N*H][D]
    __bf16* Wkt = Wqt + (size_t)N * H * D;        // [KH*H][D]
    __bf16* Wvt = Wkt + (size_t)KH * H * D;       // [KH*H][D]
    __bf16* Wot = Wvt + (size_t)KH * H * D;       // [D][N*H]

    cvt_bf16<<<(M * D / 4 + 255) / 256, 256, 0, stream>>>(X, Xb, M * D / 4);
    transpose_bf16<<<dim3((N * H) / 32, D / 32),  256, 0, stream>>>(Wq, Wqt, D, N * H);
    transpose_bf16<<<dim3((KH * H) / 32, D / 32), 256, 0, stream>>>(Wk, Wkt, D, KH * H);
    transpose_bf16<<<dim3((KH * H) / 32, D / 32), 256, 0, stream>>>(Wv, Wvt, D, KH * H);
    transpose_bf16<<<dim3(D / 32, (N * H) / 32),  256, 0, stream>>>(Wo, Wot, N * H, D);

    gemm_wmma<1><<<dim3((N * H) / TBN,  M / TBM), 256, 0, stream>>>(Xb, Wqt, Qb, M, D, N * H);
    gemm_wmma<1><<<dim3((KH * H) / TBN, M / TBM), 256, 0, stream>>>(Xb, Wkt, Kb, M, D, KH * H);
    gemm_wmma<2><<<dim3((KH * H) / TBN, M / TBM), 256, 0, stream>>>(Xb, Wvt, Vt, M, D, KH * H);

    rope_kernel<<<(M * N  * 64 + 255) / 256, 256, 0, stream>>>(Qb, qpos, M, N);
    rope_kernel<<<(M * KH * 64 + 255) / 256, 256, 0, stream>>>(Kb, qpos, M, KH);

    flash_attn<<<dim3(T / QT, N, Bn), 128, 0, stream>>>(Qb, Kb, Vt, Ab, Bn, T, N, KH);

    gemm_wmma<0><<<dim3(D / TBN, M / TBM), 256, 0, stream>>>(Ab, Wot, out, M, N * H, D);
}
